// HierarchicalHeteroGraphSage_72335839199437
// MI455X (gfx1250) — compile-verified, matrix operand order checked by reading, not run on hardware
//
#include <hip/hip_runtime.h>

#define NN 100000   // nodes
#define ET 3        // edge types
#define ME 600000   // edges per type
#define HD 128      // D == H

typedef __attribute__((ext_vector_type(16))) __bf16 v16bf;
typedef __attribute__((ext_vector_type(8)))  float  v8f;

// ---------------------------------------------------------------------------
// Fragment loader for 16x32 bf16 A/B tiles (CDNA5 WMMA layout, wave32):
//   lane&15  -> M (A) or N (B) index within tile
//   lane>>4  -> selects K half: VGPR0-3 cover K = kh..kh+7, VGPR4-7 cover K+16
// base points at tile start of a row-major [*,128] matrix; for B we pass the
// weight matrix W (out = X @ W^T), so B's column n is W's row n: same loader.
// Optional cnt pointer fuses the scatter-mean division (1/max(cnt,1)) into
// the f32->bf16 conversion of A fragments.
// ---------------------------------------------------------------------------
__device__ __forceinline__ v16bf load_frag(const float* __restrict__ base, int k0,
                                           const float* __restrict__ cnt) {
  const int lane = threadIdx.x & 31;
  const int r    = lane & 15;
  const int kh   = (lane >> 4) << 3;           // 0 or 8
  const float* p = base + r * HD + k0 + kh;
  float s = 1.0f;
  if (cnt) s = 1.0f / fmaxf(cnt[r], 1.0f);
  v16bf f;
#pragma unroll
  for (int j = 0; j < 8; ++j) f[j]     = (__bf16)(p[j]      * s);
#pragma unroll
  for (int j = 0; j < 8; ++j) f[8 + j] = (__bf16)(p[16 + j] * s);
  return f;
}

// Y = relu(X @ W^T + bias)   [NN,128] x [128,128]
__global__ __launch_bounds__(256)
void gemm_proj_relu(const float* __restrict__ X, const float* __restrict__ W,
                    const float* __restrict__ bias, float* __restrict__ Y) {
  const int row0 = blockIdx.x * 16;
  const int col0 = (threadIdx.x >> 5) * 16;    // 8 waves -> 8 col tiles
  const int lane = threadIdx.x & 31;
  const float* Ab = X + (size_t)row0 * HD;
  const float* Bb = W + (size_t)col0 * HD;
  v8f c = {};
#pragma unroll
  for (int ks = 0; ks < 4; ++ks) {
    v16bf a = load_frag(Ab, ks * 32, nullptr);
    v16bf b = load_frag(Bb, ks * 32, nullptr);
    c = __builtin_amdgcn_wmma_f32_16x16x32_bf16(false, a, false, b, (short)0, c, false, false);
  }
  const int n  = lane & 15;
  const int mh = (lane >> 4) << 3;
  const float bc = bias[col0 + n];
#pragma unroll
  for (int v = 0; v < 8; ++v)
    Y[(size_t)(row0 + mh + v) * HD + col0 + n] = fmaxf(c[v] + bc, 0.0f);
}

// Y = (AGG/max(cnt,1)) @ Wl^T + bl + X @ Wr^T    (dual GEMM, fused mean)
__global__ __launch_bounds__(256)
void gemm_combine(const float* __restrict__ AGG, const float* __restrict__ CNT,
                  const float* __restrict__ X,
                  const float* __restrict__ Wl, const float* __restrict__ bl,
                  const float* __restrict__ Wr, float* __restrict__ Y) {
  const int row0 = blockIdx.x * 16;
  const int col0 = (threadIdx.x >> 5) * 16;
  const int lane = threadIdx.x & 31;
  v8f c = {};
#pragma unroll
  for (int ks = 0; ks < 4; ++ks) {
    v16bf a = load_frag(AGG + (size_t)row0 * HD, ks * 32, CNT + row0);
    v16bf b = load_frag(Wl  + (size_t)col0 * HD, ks * 32, nullptr);
    c = __builtin_amdgcn_wmma_f32_16x16x32_bf16(false, a, false, b, (short)0, c, false, false);
  }
#pragma unroll
  for (int ks = 0; ks < 4; ++ks) {
    v16bf a = load_frag(X  + (size_t)row0 * HD, ks * 32, nullptr);
    v16bf b = load_frag(Wr + (size_t)col0 * HD, ks * 32, nullptr);
    c = __builtin_amdgcn_wmma_f32_16x16x32_bf16(false, a, false, b, (short)0, c, false, false);
  }
  const int n  = lane & 15;
  const int mh = (lane >> 4) << 3;
  const float bc = bl[col0 + n];
#pragma unroll
  for (int v = 0; v < 8; ++v)
    Y[(size_t)(row0 + mh + v) * HD + col0 + n] = c[v] + bc;
}

// one wave per edge: agg[dst] += msg[src]; cnt[dst] += 1
__global__ __launch_bounds__(256)
void scatter_add(const int* __restrict__ src, const int* __restrict__ dst,
                 const float* __restrict__ MSG, float* __restrict__ AGG,
                 float* __restrict__ CNT) {
  const int e = blockIdx.x * 8 + (threadIdx.x >> 5);
  if (e >= ME) return;
  const int lane = threadIdx.x & 31;
  const int s = src[e], d = dst[e];
  const float4 m = ((const float4*)(MSG + (size_t)s * HD))[lane];
  float* o = AGG + (size_t)d * HD + lane * 4;
  atomicAdd(o + 0, m.x);
  atomicAdd(o + 1, m.y);
  atomicAdd(o + 2, m.z);
  atomicAdd(o + 3, m.w);
  if (lane == 0) atomicAdd(CNT + d, 1.0f);
}

// one wave per row: acc += (do_norm ? T/max(||T||2, eps) : T)
__global__ __launch_bounds__(256)
void accum_rows(const float* __restrict__ T, float* __restrict__ ACC, int do_norm) {
  const int row = blockIdx.x * 8 + (threadIdx.x >> 5);
  if (row >= NN) return;
  const int lane = threadIdx.x & 31;
  float4 v = ((const float4*)(T + (size_t)row * HD))[lane];
  float s = 1.0f;
  if (do_norm) {
    float ss = v.x * v.x + v.y * v.y + v.z * v.z + v.w * v.w;
#pragma unroll
    for (int m = 16; m > 0; m >>= 1) ss += __shfl_xor(ss, m, 32);
    s = 1.0f / fmaxf(sqrtf(ss), 1e-12f);
  }
  float4* a = ((float4*)(ACC + (size_t)row * HD)) + lane;
  float4 av = *a;
  av.x += v.x * s; av.y += v.y * s; av.z += v.z * s; av.w += v.w * s;
  *a = av;
}

// one wave per row: OUT = LayerNorm(relu(ACC/3)) * gamma + beta
__global__ __launch_bounds__(256)
void finalize_rows(const float* __restrict__ ACC, const float* __restrict__ g,
                   const float* __restrict__ b, float* __restrict__ OUT) {
  const int row = blockIdx.x * 8 + (threadIdx.x >> 5);
  if (row >= NN) return;
  const int lane = threadIdx.x & 31;
  float4 v = ((const float4*)(ACC + (size_t)row * HD))[lane];
  const float inv3 = 1.0f / 3.0f;
  v.x = fmaxf(v.x * inv3, 0.0f);
  v.y = fmaxf(v.y * inv3, 0.0f);
  v.z = fmaxf(v.z * inv3, 0.0f);
  v.w = fmaxf(v.w * inv3, 0.0f);
  float sum = v.x + v.y + v.z + v.w;
#pragma unroll
  for (int m = 16; m > 0; m >>= 1) sum += __shfl_xor(sum, m, 32);
  const float mu = sum * (1.0f / 128.0f);
  const float d0 = v.x - mu, d1 = v.y - mu, d2 = v.z - mu, d3 = v.w - mu;
  float ssq = d0 * d0 + d1 * d1 + d2 * d2 + d3 * d3;
#pragma unroll
  for (int m = 16; m > 0; m >>= 1) ssq += __shfl_xor(ssq, m, 32);
  const float r = rsqrtf(ssq * (1.0f / 128.0f) + 1e-5f);
  const float4 gv = ((const float4*)g)[lane];
  const float4 bv = ((const float4*)b)[lane];
  float4 o;
  o.x = d0 * r * gv.x + bv.x;
  o.y = d1 * r * gv.y + bv.y;
  o.z = d2 * r * gv.z + bv.z;
  o.w = d3 * r * gv.w + bv.w;
  ((float4*)(OUT + (size_t)row * HD))[lane] = o;
}

__global__ __launch_bounds__(256)
void zero_f32(float* __restrict__ p, int n4) {
  const int i = blockIdx.x * 256 + threadIdx.x;
  if (i < n4) ((float4*)p)[i] = make_float4(0.f, 0.f, 0.f, 0.f);
}

extern "C" void kernel_launch(void* const* d_in, const int* in_sizes, int n_in,
                              void* d_out, int out_size, void* d_ws, size_t ws_size,
                              hipStream_t stream) {
  const float* x      = (const float*)d_in[0];
  const int*   ei     = (const int*)d_in[1];
  // d_in[2], d_in[3]: neighbor masks — unused by the reference
  const float* proj_W = (const float*)d_in[4];
  const float* proj_b = (const float*)d_in[5];
  const float* l0_Wl  = (const float*)d_in[6];
  const float* l0_bl  = (const float*)d_in[7];
  const float* l0_Wr  = (const float*)d_in[8];
  const float* l1_Wl  = (const float*)d_in[9];
  const float* l1_bl  = (const float*)d_in[10];
  const float* l1_Wr  = (const float*)d_in[11];
  const float* ln_g   = (const float*)d_in[12];
  const float* ln_b   = (const float*)d_in[13];
  float* outp = (float*)d_out;

  const size_t NH = (size_t)NN * HD;
  float* tmp = (float*)d_ws;        // projected messages / combine output (reused)
  float* agg = tmp + NH;            // scatter accumulator
  float* cnt = agg + NH;            // degree counts (adjacent to agg: zeroed together)
  float* acc = cnt + NN;            // per-layer hetero-mean accumulator
  float* h0  = outp;                // layer-0 hidden staged in d_out (dead before final LN)

  const dim3 blk(256);
  const dim3 ggemm(NN / 16);                         // 6250
  const dim3 gedge(ME / 8);                          // 75000
  const dim3 grow(NN / 8);                           // 12500
  const dim3 gz_aggcnt((int)((NH + NN) / 4 + 255) / 256);
  const dim3 gz_acc((int)(NH / 4 + 255) / 256);

  // ---------------- Layer 0: project=True, normalize=True ----------------
  zero_f32<<<gz_acc, blk, 0, stream>>>(acc, (int)(NH / 4));
  for (int e = 0; e < ET; ++e) {
    const int* src = ei + (size_t)e * 2 * ME;
    const int* dst = src + ME;
    gemm_proj_relu<<<ggemm, blk, 0, stream>>>(x, proj_W + (size_t)e * HD * HD,
                                              proj_b + e * HD, tmp);
    zero_f32<<<gz_aggcnt, blk, 0, stream>>>(agg, (int)((NH + NN) / 4));
    scatter_add<<<gedge, blk, 0, stream>>>(src, dst, tmp, agg, cnt);
    gemm_combine<<<ggemm, blk, 0, stream>>>(agg, cnt, x,
                                            l0_Wl + (size_t)e * HD * HD, l0_bl + e * HD,
                                            l0_Wr + (size_t)e * HD * HD, tmp);
    accum_rows<<<grow, blk, 0, stream>>>(tmp, acc, 1);
  }
  finalize_rows<<<grow, blk, 0, stream>>>(acc, ln_g, ln_b, h0);

  // ---------------- Layer 1: plain SAGEConv ----------------
  zero_f32<<<gz_acc, blk, 0, stream>>>(acc, (int)(NH / 4));
  for (int e = 0; e < ET; ++e) {
    const int* src = ei + (size_t)e * 2 * ME;
    const int* dst = src + ME;
    zero_f32<<<gz_aggcnt, blk, 0, stream>>>(agg, (int)((NH + NN) / 4));
    scatter_add<<<gedge, blk, 0, stream>>>(src, dst, h0, agg, cnt);
    gemm_combine<<<ggemm, blk, 0, stream>>>(agg, cnt, h0,
                                            l1_Wl + (size_t)e * HD * HD, l1_bl + e * HD,
                                            l1_Wr + (size_t)e * HD * HD, tmp);
    accum_rows<<<grow, blk, 0, stream>>>(tmp, acc, 0);
  }
  finalize_rows<<<grow, blk, 0, stream>>>(acc, ln_g + HD, ln_b + HD, outp);
}
